// Attention_30562987278646
// MI455X (gfx1250) — compile-verified, matrix operand order checked by reading, not run on hardware
//
#include <hip/hip_runtime.h>
#include <hip/hip_bf16.h>
#include <math.h>

// ---------------- types ----------------
typedef _Float16 v8h  __attribute__((ext_vector_type(8)));
typedef _Float16 v16h __attribute__((ext_vector_type(16)));
typedef float    v8f  __attribute__((ext_vector_type(8)));

#define BB    4
#define NN    1024
#define DIMX  512
#define HEADS 8
#define DH    64
#define MTOT  4096   // B*N

// ---------------- WMMA helpers ----------------
static __device__ inline v8f wmma_f16(v16h a, v16h b, v8f c) {
  // D = A(16x32 f16) x B(32x16 f16) + C(16x16 f32)
  return __builtin_amdgcn_wmma_f32_16x16x32_f16(
      /*neg_a=*/false, a, /*neg_b=*/false, b,
      /*c_mod=*/(short)0, c, /*reuse_a=*/false, /*reuse_b=*/false);
}

// Build a 16-half A/B fragment from two contiguous 8-half (16B) chunks.
// CDNA5 16-bit layout: lane<16 -> K{base..base+7},{base+16..base+23},
// lane>=16 -> base=8. Both chunks are 16B-aligned contiguous loads.
static __device__ inline v16h ld_frag(const _Float16* p0, const _Float16* p1) {
  v8h lo = *(const v8h*)p0;
  v8h hi = *(const v8h*)p1;
  v16h r;
#pragma unroll
  for (int i = 0; i < 8; ++i) { r[i] = lo[i]; r[8 + i] = hi[i]; }
  return r;
}

// ---------------- 1. LayerNorm (f32 in, f16 normalized out) ----------------
__global__ __launch_bounds__(256) void ln_f16_kernel(const float* __restrict__ x,
                                                     const float* __restrict__ g,
                                                     const float* __restrict__ bt,
                                                     _Float16* __restrict__ y) {
  __shared__ float rs[256], rs2[256];
  const int row = blockIdx.x, t = threadIdx.x;
  const float* xr = x + (size_t)row * DIMX;
  float s = 0.f, s2 = 0.f;
  for (int i = t; i < DIMX; i += 256) { float v = xr[i]; s += v; s2 += v * v; }
  rs[t] = s; rs2[t] = s2; __syncthreads();
  for (int o = 128; o > 0; o >>= 1) {
    if (t < o) { rs[t] += rs[t + o]; rs2[t] += rs2[t + o]; }
    __syncthreads();
  }
  const float mean = rs[0] * (1.f / DIMX);
  const float var  = rs2[0] * (1.f / DIMX) - mean * mean;
  const float inv  = rsqrtf(var + 1e-5f);
  _Float16* yr = y + (size_t)row * DIMX;
  for (int i = t; i < DIMX; i += 256) {
    float v = xr[i];
    yr[i] = (_Float16)(((v - mean) * inv) * g[i] + bt[i]);
  }
}

// ---------------- 2. f32 -> f16 convert ----------------
__global__ __launch_bounds__(256) void cvt_f16_kernel(const float* __restrict__ s,
                                                      _Float16* __restrict__ d, int n) {
  int i = blockIdx.x * 256 + threadIdx.x;
  if (i < n) d[i] = (_Float16)s[i];
}

// ---------------- 3. Tiled WMMA GEMM: C[M,N] = A[M,K]f16 * B[K,N]f16 (+bias) --------
// 256 threads = 8 waves; block tile 128(M) x 64(N); K chunk 32.
// Each wave owns a 16x64 strip: A fragment reused across 4 WMMA per chunk.
// All fragment loads are hoisted ahead of the WMMA group so the four
// v_wmma issue back-to-back after a single s_wait_dscnt.
__global__ __launch_bounds__(256) void gemm_wmma_kernel(const _Float16* __restrict__ A,
                                                        const _Float16* __restrict__ Bm,
                                                        float* __restrict__ Cf32,
                                                        _Float16* __restrict__ Cf16,
                                                        const float* __restrict__ bias,
                                                        int Mdim, int Ndim, int Kdim) {
  __shared__ __align__(16) _Float16 As[128 * 32];   // row-major  [m][k]   (8 KiB)
  __shared__ __align__(16) _Float16 Bts[64 * 32];   // transposed [n][k]   (4 KiB)
  const int t = threadIdx.x;
  const int wave = t >> 5, lane = t & 31, lm = lane & 15, half = lane >> 4;
  const int m0 = blockIdx.x * 128, n0 = blockIdx.y * 64;

  v8f acc0 = {0,0,0,0,0,0,0,0}, acc1 = {0,0,0,0,0,0,0,0};
  v8f acc2 = {0,0,0,0,0,0,0,0}, acc3 = {0,0,0,0,0,0,0,0};

  // staging index precompute
  const int ar = t >> 1, ac = (t & 1) * 16;         // A: 16 halves per thread
  const int bk = t >> 3, bs = (t & 7) * 8;          // B: one 16B row chunk per thread

  for (int k0 = 0; k0 < Kdim; k0 += 32) {
    // stage A tile: two 16B vector copies per thread
    {
      const _Float16* ag = A + (size_t)(m0 + ar) * Kdim + k0 + ac;
      *(v8h*)&As[ar * 32 + ac]     = *(const v8h*)(ag);
      *(v8h*)&As[ar * 32 + ac + 8] = *(const v8h*)(ag + 8);
      __builtin_prefetch(ag + 32, 0, 1);            // next K tile (global_prefetch_b8)
    }
    // stage B tile transposed: one b128 global read, 8 transposed b16 stores
    {
      const _Float16* bg = Bm + (size_t)(k0 + bk) * Ndim + n0 + bs;
      v8h bch = *(const v8h*)bg;
      __builtin_prefetch(bg + (size_t)32 * Ndim, 0, 1);
#pragma unroll
      for (int j = 0; j < 8; ++j) Bts[(bs + j) * 32 + bk] = bch[j];
    }
    __syncthreads();

    const int koff = half * 8;
    const int arow = wave * 16 + lm;
    // hoist ALL fragment loads before the WMMA group
    v16h af = ld_frag(&As[arow * 32 + koff], &As[arow * 32 + koff + 16]);
    v16h b0 = ld_frag(&Bts[(0 * 16 + lm) * 32 + koff], &Bts[(0 * 16 + lm) * 32 + koff + 16]);
    v16h b1 = ld_frag(&Bts[(1 * 16 + lm) * 32 + koff], &Bts[(1 * 16 + lm) * 32 + koff + 16]);
    v16h b2 = ld_frag(&Bts[(2 * 16 + lm) * 32 + koff], &Bts[(2 * 16 + lm) * 32 + koff + 16]);
    v16h b3 = ld_frag(&Bts[(3 * 16 + lm) * 32 + koff], &Bts[(3 * 16 + lm) * 32 + koff + 16]);
    acc0 = wmma_f16(af, b0, acc0);
    acc1 = wmma_f16(af, b1, acc1);
    acc2 = wmma_f16(af, b2, acc2);
    acc3 = wmma_f16(af, b3, acc3);
    __syncthreads();
  }

  // epilogue: f32 C/D layout: row = v + (lane>=16?8:0), col = lane%16
  const int row = m0 + wave * 16 + (half ? 8 : 0);
#pragma unroll
  for (int c = 0; c < 4; ++c) {
    const int col = n0 + c * 16 + lm;
    const float bv = bias ? bias[col] : 0.f;
    const v8f* accp = (c == 0) ? &acc0 : (c == 1) ? &acc1 : (c == 2) ? &acc2 : &acc3;
#pragma unroll
    for (int v = 0; v < 8; ++v) {
      const float val = (*accp)[v] + bv;
      const size_t idx = (size_t)(row + v) * Ndim + col;
      if (Cf32) Cf32[idx] = val;
      else      Cf16[idx] = (_Float16)val;
    }
  }
}

// ---------------- 4. per-(h,b,n) stats: mean & 1/norm over d ----------------
__global__ __launch_bounds__(256) void stats_kernel(const _Float16* __restrict__ F,
                                                    float* __restrict__ mu,
                                                    float* __restrict__ invn) {
  const int idx = blockIdx.x * 256 + threadIdx.x;
  if (idx >= MTOT * HEADS) return;
  const int row = idx >> 3, h = idx & 7;
  const _Float16* p = F + (size_t)row * DIMX + h * DH;
  float s = 0.f, s2 = 0.f;
#pragma unroll 8
  for (int i = 0; i < DH; ++i) { float v = (float)p[i]; s += v; s2 += v * v; }
  const int b = row >> 10, n = row & 1023;
  const int o = ((h << 2) + b) * NN + n;
  mu[o]   = s * (1.f / DH);
  invn[o] = 1.f / fmaxf(sqrtf(s2), 1e-12f);
}

// ---------------- 5. colsum of f_v per (h,b) ----------------
__global__ __launch_bounds__(64) void colsumv_kernel(const _Float16* __restrict__ Fv,
                                                     float* __restrict__ colV) {
  const int hb = blockIdx.x;                 // h*4+b
  const int h = hb >> 2, b = hb & 3, d = threadIdx.x;
  float s = 0.f;
  for (int n = 0; n < NN; ++n)
    s += (float)Fv[(size_t)(b * NN + n) * DIMX + h * DH + d];
  colV[hb * DH + d] = s;
}

// ---------------- 6. attention: 4 waves/block, same (h,b), shared Vt ----------------
__global__ __launch_bounds__(128) void attn_wmma_kernel(
    const _Float16* __restrict__ Fq, const _Float16* __restrict__ Fk,
    const _Float16* __restrict__ Fv,
    const float* __restrict__ muq, const float* __restrict__ invq,
    const float* __restrict__ muk, const float* __restrict__ invk,
    const float* __restrict__ colV,
    const float* __restrict__ cov_raw, const float* __restrict__ var_raw,
    _Float16* __restrict__ AttnO) {
  __shared__ __align__(16) _Float16 Vt[DH * 32];       // [d][m] transposed V chunk (shared)
  __shared__ __align__(16) _Float16 Ss[4][16 * 32];    // per-wave score staging

  const int t = threadIdx.x;
  const int wave = t >> 5, lane = t & 31, lm = lane & 15, half = lane >> 4;
  const int rowoff = half ? 8 : 0;
  const int id = blockIdx.x;
  const int ntg = id & 15, hb = id >> 4;               // 16 tile-groups per (h,b)
  const int b = hb & 3, h = hb >> 2;
  const int n0 = (ntg * 4 + wave) * 16;                // this wave's 16-row tile
  const int hb1024 = hb * NN;

  const float cov_w = 1.f / (1.f + expf(-cov_raw[0]));
  const float var_w = 1.f / (1.f + expf(-var_raw[0]));
  const float cos_w = 1.f - cov_w - var_w;

  // Q A-fragments directly from global (row-major matches A layout)
  const _Float16* qbase = Fq + (size_t)(b * NN + n0 + lm) * DIMX + h * DH;
  const v16h aq0 = ld_frag(qbase + half * 8,      qbase + half * 8 + 16);
  const v16h aq1 = ld_frag(qbase + 32 + half * 8, qbase + 32 + half * 8 + 16);

  float invq_r[8], muq_r[8];
#pragma unroll
  for (int v = 0; v < 8; ++v) {
    const int r = n0 + v + rowoff;
    invq_r[v] = invq[hb1024 + r];
    muq_r[v]  = muq[hb1024 + r];
  }

  v8f accO0 = {0,0,0,0,0,0,0,0}, accO1 = {0,0,0,0,0,0,0,0};
  v8f accO2 = {0,0,0,0,0,0,0,0}, accO3 = {0,0,0,0,0,0,0,0};
  float vsum[8] = {0,0,0,0,0,0,0,0};

  // Vt staging split: 128 threads, 4 per m-row, 16 d-elements each
  const int smm = t >> 2, sseg = (t & 3) * 16;

  for (int m0 = 0; m0 < NN; m0 += 32) {
    // cooperative stage of the (shared) transposed V chunk: 2x b128 global reads
    {
      const _Float16* vrow = Fv + (size_t)(b * NN + m0 + smm) * DIMX + h * DH + sseg;
      v8h c0 = *(const v8h*)(vrow);
      v8h c1 = *(const v8h*)(vrow + 8);
#pragma unroll
      for (int j = 0; j < 8; ++j) {
        Vt[(sseg + j) * 32 + smm]     = c0[j];
        Vt[(sseg + 8 + j) * 32 + smm] = c1[j];
      }
    }
    __syncthreads();   // Vt ready for all waves

    // dots = Q @ K^T : two 16x16 m-subtiles, K-dim = d = 64 (2 chained wmma each).
    // B = K^T -> column n of B is a row of K: contiguous global loads.
#pragma unroll
    for (int j = 0; j < 2; ++j) {
      const int mrow = m0 + j * 16 + lm;
      const _Float16* kb = Fk + (size_t)(b * NN + mrow) * DIMX + h * DH;
      v16h bk0 = ld_frag(kb + half * 8,      kb + half * 8 + 16);
      v16h bk1 = ld_frag(kb + 32 + half * 8, kb + 32 + half * 8 + 16);
      v8f dfr = {0,0,0,0,0,0,0,0};
      dfr = wmma_f16(aq0, bk0, dfr);
      dfr = wmma_f16(aq1, bk1, dfr);

      const float ik = invk[hb1024 + mrow];
      const float mk = muk[hb1024 + mrow];
#pragma unroll
      for (int v = 0; v < 8; ++v) {
        const float dot  = dfr[v];
        const float cosv = dot * invq_r[v] * ik;
        const float covv = (dot - 64.f * muq_r[v] * mk) * (1.f / 64.f);
        vsum[v] += fmaxf(1.f - cosv, 0.f);                    // GAMMA = 1
        Ss[wave][(v + rowoff) * 32 + j * 16 + lm] =
            (_Float16)(cos_w * cosv + cov_w * covv);
      }
    }
    __syncthreads();   // fence score stores before fragment reloads

    // S @ V : A = S[16x32] (K = m), B = V[32x64] via shared transposed Vt.
    // Hoist all 5 fragment loads so the 4 WMMA issue back-to-back.
    v16h as  = ld_frag(&Ss[wave][lm * 32 + half * 8], &Ss[wave][lm * 32 + half * 8 + 16]);
    v16h bv0 = ld_frag(&Vt[(0 * 16 + lm) * 32 + half * 8], &Vt[(0 * 16 + lm) * 32 + half * 8 + 16]);
    v16h bv1 = ld_frag(&Vt[(1 * 16 + lm) * 32 + half * 8], &Vt[(1 * 16 + lm) * 32 + half * 8 + 16]);
    v16h bv2 = ld_frag(&Vt[(2 * 16 + lm) * 32 + half * 8], &Vt[(2 * 16 + lm) * 32 + half * 8 + 16]);
    v16h bv3 = ld_frag(&Vt[(3 * 16 + lm) * 32 + half * 8], &Vt[(3 * 16 + lm) * 32 + half * 8 + 16]);
    accO0 = wmma_f16(as, bv0, accO0);
    accO1 = wmma_f16(as, bv1, accO1);
    accO2 = wmma_f16(as, bv2, accO2);
    accO3 = wmma_f16(as, bv3, accO3);
    __syncthreads();   // all waves done with Vt before next chunk overwrites it
  }

  // reduce margin sums across the 16 lanes of each half (rows 0-7 in lanes 0-15,
  // rows 8-15 in lanes 16-31; xor offsets 1,2,4,8 stay within a half)
#pragma unroll
  for (int v = 0; v < 8; ++v) {
    float s = vsum[v];
    for (int off = 1; off < 16; off <<= 1) s += __shfl_xor(s, off, 32);
    vsum[v] = s * (1.f / NN) * var_w;     // var_w * mean_m(relu(1-cos))
  }

  const float* cvp = colV + hb * DH;
#pragma unroll
  for (int v = 0; v < 8; ++v) {
    const size_t gr = (size_t)(b * NN + n0 + v + rowoff);
    _Float16* orow = AttnO + gr * DIMX + h * DH;
    const float vr = vsum[v];
    orow[0 * 16 + lm] = (_Float16)(accO0[v] + vr * cvp[0 * 16 + lm]);
    orow[1 * 16 + lm] = (_Float16)(accO1[v] + vr * cvp[1 * 16 + lm]);
    orow[2 * 16 + lm] = (_Float16)(accO2[v] + vr * cvp[2 * 16 + lm]);
    orow[3 * 16 + lm] = (_Float16)(accO3[v] + vr * cvp[3 * 16 + lm]);
  }
}

// ---------------- launcher ----------------
extern "C" void kernel_launch(void* const* d_in, const int* in_sizes, int n_in,
                              void* d_out, int out_size, void* d_ws, size_t ws_size,
                              hipStream_t stream) {
  const float* q        = (const float*)d_in[0];
  const float* k        = (const float*)d_in[1];
  const float* v        = (const float*)d_in[2];
  const float* ln_g     = (const float*)d_in[3];
  const float* ln_b     = (const float*)d_in[4];
  const float* W_in     = (const float*)d_in[5];
  const float* W_out    = (const float*)d_in[6];
  const float* b_out    = (const float*)d_in[7];
  const float* cov_raw  = (const float*)d_in[8];
  const float* var_raw  = (const float*)d_in[9];
  float* out = (float*)d_out;

  // workspace layout
  char* w = (char*)d_ws;
  const size_t tensH = (size_t)MTOT * DIMX * sizeof(_Float16);  // 4 MiB
  const size_t wH    = (size_t)DIMX * DIMX * sizeof(_Float16);  // 512 KiB
  const size_t statF = (size_t)HEADS * BB * NN * sizeof(float); // 128 KiB
  _Float16* Xq   = (_Float16*)w; w += tensH;
  _Float16* Xk   = (_Float16*)w; w += tensH;
  _Float16* Xv   = (_Float16*)w; w += tensH;
  _Float16* Wi16 = (_Float16*)w; w += wH;
  _Float16* Wo16 = (_Float16*)w; w += wH;
  _Float16* Fq   = (_Float16*)w; w += tensH;
  _Float16* Fk   = (_Float16*)w; w += tensH;
  _Float16* Fv   = (_Float16*)w; w += tensH;
  _Float16* AtO  = (_Float16*)w; w += tensH;
  float* muqP  = (float*)w; w += statF;
  float* invqP = (float*)w; w += statF;
  float* mukP  = (float*)w; w += statF;
  float* invkP = (float*)w; w += statF;
  float* colV  = (float*)w; w += (size_t)HEADS * BB * DH * sizeof(float);

  // 1) LayerNorm + f16
  ln_f16_kernel<<<MTOT, 256, 0, stream>>>(q, ln_g, ln_b, Xq);
  ln_f16_kernel<<<MTOT, 256, 0, stream>>>(k, ln_g, ln_b, Xk);
  ln_f16_kernel<<<MTOT, 256, 0, stream>>>(v, ln_g, ln_b, Xv);

  // 2) weights to f16
  cvt_f16_kernel<<<(DIMX * DIMX + 255) / 256, 256, 0, stream>>>(W_in, Wi16, DIMX * DIMX);
  cvt_f16_kernel<<<(DIMX * DIMX + 255) / 256, 256, 0, stream>>>(W_out, Wo16, DIMX * DIMX);

  // 3) projection GEMMs (WMMA), f16 outputs; block tile 128x64
  dim3 gg(MTOT / 128, DIMX / 64);
  gemm_wmma_kernel<<<gg, 256, 0, stream>>>(Xq, Wi16, nullptr, Fq, nullptr, MTOT, DIMX, DIMX);
  gemm_wmma_kernel<<<gg, 256, 0, stream>>>(Xk, Wi16, nullptr, Fk, nullptr, MTOT, DIMX, DIMX);
  gemm_wmma_kernel<<<gg, 256, 0, stream>>>(Xv, Wi16, nullptr, Fv, nullptr, MTOT, DIMX, DIMX);

  // 4) per-(h,b,n) stats for q,k
  stats_kernel<<<(MTOT * HEADS + 255) / 256, 256, 0, stream>>>(Fq, muqP, invqP);
  stats_kernel<<<(MTOT * HEADS + 255) / 256, 256, 0, stream>>>(Fk, mukP, invkP);

  // 5) colsum(f_v) per (h,b)
  colsumv_kernel<<<HEADS * BB, 64, 0, stream>>>(Fv, colV);

  // 6) attention (WMMA): 4 waves/block sharing the transposed V chunk
  attn_wmma_kernel<<<HEADS * BB * 16, 128, 0, stream>>>(
      Fq, Fk, Fv, muqP, invqP, mukP, invkP, colV, cov_raw, var_raw, AtO);

  // 7) output GEMM (WMMA) + bias -> f32
  gemm_wmma_kernel<<<gg, 256, 0, stream>>>(AtO, Wo16, out, nullptr, b_out, MTOT, DIMX, DIMX);
}
